// ACMELinear_90048284328685
// MI455X (gfx1250) — matrix operand order repaired and checked by reading, not compiled, not run on hardware
//
#include <hip/hip_runtime.h>
#include <hip/hip_bf16.h>

typedef __attribute__((ext_vector_type(16))) _Float16 v16h;
typedef __attribute__((ext_vector_type(8)))  _Float16 v8h;
typedef __attribute__((ext_vector_type(8)))  float    v8f;

#define M_DIM 128
#define N_DIM 8192
#define K_DIM 8192
#define N_STRIPS (N_DIM / 16)      // 512 N strips of width 16
#define KSPLIT 4
#define KCHUNK (K_DIM / KSPLIT)    // 2048
#define SLABS  (KCHUNK / 32)       // 64 K-slabs of 32 per wave
#define MASK_W (K_DIM / 64)        // 128 mask tiles per row
#define TILES_PER_CHUNK (KCHUNK / 64)  // 32 mask tiles per K-chunk (== wave32 lanes)
#define LDS_ROW 40                 // halves per LDS row (80 B, bank-conflict-free pad)

// ---- async global -> LDS 16-byte copy (gfx1250 ASYNC path, ASYNCcnt) ----
__device__ __forceinline__ void async_copy16(const _Float16* g, _Float16* l) {
    const unsigned loff = (unsigned)(unsigned long long)
        (__attribute__((address_space(3))) _Float16*)l;
    asm volatile("global_load_async_to_lds_b128 %0, %1, off"
                 :: "v"(loff), "v"(g) : "memory");
}

__device__ __forceinline__ void wait_async0() {
#if __has_builtin(__builtin_amdgcn_s_wait_asynccnt)
    __builtin_amdgcn_s_wait_asynccnt(0);
#else
    asm volatile("s_wait_asynccnt 0x0" ::: "memory");
#endif
}

__device__ __forceinline__ unsigned ballot32(bool p) {
#if __has_builtin(__builtin_amdgcn_ballot_w32)
    return __builtin_amdgcn_ballot_w32(p);
#else
    return (unsigned)__ballot((int)p);
#endif
}

// ---- pre-pass 1: x fp32 -> f16, row-major [M, K] into workspace ----
__global__ __launch_bounds__(256) void xcvt_kernel(const float* __restrict__ x,
                                                   _Float16* __restrict__ xh, int n) {
    int i = blockIdx.x * blockDim.x + threadIdx.x;
    if (i < n) xh[i] = (_Float16)x[i];
}

// ---- pre-pass 2: out[m,n] = bias[n] (GEMM reduces into it with atomics) ----
__global__ __launch_bounds__(256) void binit_kernel(const float* __restrict__ bias,
                                                    float* __restrict__ out, int n) {
    int i = blockIdx.x * blockDim.x + threadIdx.x;
    if (i < n) out[i] = bias[i & (N_DIM - 1)];
}

// ---- main ternary-weight GEMM: D = x * T(wb+wt)^T, f16 WMMA, f32 accum ----
__global__ __launch_bounds__(256) void ternary_gemm_kernel(
    const _Float16* __restrict__ xh,
    const float* __restrict__ wb,
    const float* __restrict__ wt,
    const float* __restrict__ tmask,
    float* __restrict__ out) {

    // double-buffered x tile: [2][128 rows][40 halves] = 20 KB
    __shared__ _Float16 lbuf[2][M_DIM * LDS_ROW];

    const int tid    = threadIdx.x;
    const int lane   = tid & 31;
    const int gwave  = blockIdx.x * (blockDim.x >> 5) + (tid >> 5);
    const int nstrip = gwave & (N_STRIPS - 1);        // which 16-wide N strip
    const int kchunk = gwave >> 9;                    // which K partition
    const int nbase  = nstrip << 4;
    const int k0     = kchunk * KCHUNK;

    // A-fragment lane geometry (16-bit A 16x32 layout)
    const int arow  = lane & 15;                      // M row within tile
    const int aoff0 = (lane >> 4) << 3;               // K: 0..7 (lo) / 8..15 (hi)
    const int maskRow = (nbase >> 6) * MASK_W;

    // ---- preload this K-chunk's 32 mask tiles: 1 per lane -> uniform bitmask ----
    const float mv = tmask[maskRow + kchunk * TILES_PER_CHUNK + lane];
    const unsigned mbits = ballot32(mv != 0.0f);      // bit t = tile t non-zero

    v8f acc[8] = {};

    // prologue: stage slab 0 (512 16B chunks, 2 per thread)
    {
#pragma unroll
        for (int j = 0; j < 2; ++j) {
            const int c   = tid + 256 * j;
            const int row = c >> 2, ci = c & 3;
            async_copy16(xh + (size_t)row * K_DIM + k0 + ci * 8,
                         &lbuf[0][row * LDS_ROW + ci * 8]);
        }
        wait_async0();
        __syncthreads();
    }

    for (int s = 0; s < SLABS; ++s) {
        const int kb  = k0 + (s << 5);
        const int cur = s & 1;

        // issue async staging of next slab into the other buffer
        if (s + 1 < SLABS) {
#pragma unroll
            for (int j = 0; j < 2; ++j) {
                const int c   = tid + 256 * j;
                const int row = c >> 2, ci = c & 3;
                async_copy16(xh + (size_t)row * K_DIM + (kb + 32) + ci * 8,
                             &lbuf[cur ^ 1][row * LDS_ROW + ci * 8]);
            }
        }

        // scalar mask-skip: tile index within chunk = s/2 (two slabs per 64-wide tile)
        if ((mbits >> (s >> 1)) & 1u) {
            // ---- issue all 8 A-fragment LDS loads first (distinct regs) ----
            v16h af[8];
#pragma unroll
            for (int t = 0; t < 8; ++t) {
                const _Float16* ap =
                    &lbuf[cur][(t * 16 + arow) * LDS_ROW + aoff0];
                const v8h lo = *(const v8h*)(ap);
                const v8h hi = *(const v8h*)(ap + 16);
#pragma unroll
                for (int i = 0; i < 8; ++i) { af[t][i] = lo[i]; af[t][i + 8] = hi[i]; }
            }

            // ---- build ternary B fragment (K=32 x N=16), {-1,0,1} in f16 ----
            const int kl = kb + ((lane >> 4) << 3);   // B layout: N = lane&15, K = kb + 8h + s (+8 for s>=8)
            v16h bf;
#pragma unroll
            for (int n = 0; n < 16; ++n) {
                const size_t idx = (size_t)(nbase + (lane & 15)) * K_DIM + kl + n + ((n >> 3) << 3);
                const float v  = wb[idx] + wt[idx];
                const float av = __builtin_fabsf(v);
                const float r  = (av >= 0.3f) ? ((v < 0.0f) ? -1.0f : 1.0f) : 0.0f;
                bf[n] = (_Float16)r;
            }

            // prefetch next slab's weight lines
            if (s + 1 < SLABS) {
                const size_t pidx = (size_t)(nbase + arow) * K_DIM + (kb + 32)
                                    + ((size_t)(lane >> 4) << 4);
                __builtin_prefetch(wb + pidx, 0, 1);
                __builtin_prefetch(wt + pidx, 0, 1);
            }

            // ---- 8 back-to-back WMMAs: all 128 batch rows vs this B fragment ----
#pragma unroll
            for (int t = 0; t < 8; ++t) {
                acc[t] = __builtin_amdgcn_wmma_f32_16x16x32_f16(
                    false, af[t], false, bf, (short)0, acc[t], false, false);
            }
        }

        // make next-slab staging visible to all waves before consuming it
        wait_async0();
        __syncthreads();
    }

    // ---- epilogue: C/D layout (VGPR r, lane): M = r + 8*(lane>=16), N = lane&15 ----
    const int ncol = nbase + (lane & 15);
    const int mhi  = (lane >> 4) << 3;
#pragma unroll
    for (int t = 0; t < 8; ++t) {
#pragma unroll
        for (int r = 0; r < 8; ++r) {
            const int m = t * 16 + mhi + r;
            atomicAdd(out + (size_t)m * N_DIM + ncol, acc[t][r]);
        }
    }
}

extern "C" void kernel_launch(void* const* d_in, const int* in_sizes, int n_in,
                              void* d_out, int out_size, void* d_ws, size_t ws_size,
                              hipStream_t stream) {
    const float* x    = (const float*)d_in[0];   // [128, 8192]
    const float* wb   = (const float*)d_in[1];   // [8192, 8192]
    const float* wt   = (const float*)d_in[2];   // [8192, 8192]
    const float* tm   = (const float*)d_in[3];   // [128, 128]
    const float* bias = (const float*)d_in[4];   // [8192]
    float* out = (float*)d_out;                  // [128, 8192]
    _Float16* xh = (_Float16*)d_ws;              // 2 MB f16 copy of x

    const int xn = M_DIM * K_DIM;
    const int on = M_DIM * N_DIM;
    xcvt_kernel<<<(xn + 255) / 256, 256, 0, stream>>>(x, xh, xn);
    binit_kernel<<<(on + 255) / 256, 256, 0, stream>>>(bias, out, on);

    // 512 N-strips x 4 K-partitions = 2048 waves -> 256 blocks of 8 waves
    const int nblocks = (N_STRIPS * KSPLIT * 32) / 256;
    ternary_gemm_kernel<<<nblocks, 256, 0, stream>>>(xh, wb, wt, tm, out);
}
